// FairDGE_13039520710733
// MI455X (gfx1250) — matrix operand order repaired and checked by reading, not compiled, not run on hardware
//
#include <hip/hip_runtime.h>

// ---------------------------------------------------------------------------
// FairDGE forward on gfx1250 (MI455X): 2-layer sparse GAT per timestep,
// two GRUs, fusion + classifier. Dense layers via v_wmma_f32_16x16x32_bf16,
// A-tiles staged to LDS via the Tensor Data Mover (tensor_load_to_lds).
// ---------------------------------------------------------------------------

typedef __attribute__((ext_vector_type(16))) __bf16 v16bf;
typedef __attribute__((ext_vector_type(8)))  float  v8f;
typedef __attribute__((ext_vector_type(4)))  unsigned int u32x4;
typedef __attribute__((ext_vector_type(8)))  unsigned int u32x8;

#define USE_TDM 1

#define TT 8
#define NN 10000
#define EE 320000
#define DD 128
#define HH 4

#define GBM 64
#define GBN 128
#define GBK 32
#define LDSK 40   // padded K-stride (40 el = 80B, keeps 16B alignment)

__device__ __forceinline__ float b2f(unsigned short u) {
    return __uint_as_float(((unsigned)u) << 16);
}
__device__ __forceinline__ float sigmoidf_(float x) {
    return 1.0f / (1.0f + expf(-x));
}

#if USE_TDM
// ---------------------------------------------------------------------------
// Tensor Data Mover: DMA a 2D bf16 tile (tile_d1 rows x tile_d0 cols, row
// stride = row_stride elements) from global to LDS. tensor_d0/d1 are the
// remaining tensor extents measured from the tile start (hardware zero-fills
// OOB reads -> replaces tail-row guards). pad: after every 16 DWORDs (one
// 64B row) insert 4 DWORDs (8 bf16) -> matches LDSK=40 padded layout.
// D# layout per CDNA5 ISA (2-group form, 2D).
// ---------------------------------------------------------------------------
__device__ __forceinline__ void tdm_load_2d_bf16(
    const void* gsrc, unsigned lds_byte_addr,
    unsigned tensor_d0, unsigned tensor_d1,
    unsigned tile_d0, unsigned tile_d1,
    unsigned row_stride, unsigned pad_interval, unsigned pad_amount)
{
    unsigned long long ga = (unsigned long long)gsrc;
    u32x4 g0;
    g0[0] = 1u;                                   // count=1, user descriptor
    g0[1] = lds_byte_addr;                        // [63:32] lds_addr
    g0[2] = (unsigned)(ga & 0xffffffffu);         // [95:64] global_addr lo
    g0[3] = (unsigned)((ga >> 32) & 0x1ffffffu)   // [120:96] global_addr hi
          | (2u << 30);                           // [127:126] type=2 (image)
    u32x8 g1;
    g1[0] = (1u << 16)                            // data_size = 2 bytes
          | (1u << 20)                            // pad_enable
          | ((pad_interval & 7u) << 22)           // pad_interval
          | ((pad_amount & 127u) << 25);          // pad_amount
    g1[1] = (tensor_d0 & 0xffffu) << 16;          // [63:48] tensor_dim0 lo
    g1[2] = ((tensor_d0 >> 16) & 0xffffu)         // [79:64] tensor_dim0 hi
          | ((tensor_d1 & 0xffffu) << 16);        // [95:80] tensor_dim1 lo
    g1[3] = ((tensor_d1 >> 16) & 0xffffu)         // [111:96] tensor_dim1 hi
          | ((tile_d0 & 0xffffu) << 16);          // [127:112] tile_dim0
    g1[4] = (tile_d1 & 0xffffu);                  // [143:128] tile_dim1 (dim2=0)
    g1[5] = row_stride;                           // [191:160] dim0_stride lo
    g1[6] = 0u;                                   // stride hi, dim1_stride lo
    g1[7] = 0u;
    asm volatile("tensor_load_to_lds %0, %1" :: "s"(g0), "s"(g1) : "memory");
}
#endif

// ---------------------------------------------------------------------------
// Tiled bf16 GEMM: C[M,Nc] = A[M,K] * B[K,Nc] (+bias) (+relu)
// 256 threads = 8 waves; waves in 4x2 grid; each wave -> 16 rows x 64 cols
// (four 16x16 WMMA tiles). A staged via TDM; B transposed into LDS.
// ---------------------------------------------------------------------------
__global__ __launch_bounds__(256) void k_gemm(
    const __bf16* __restrict__ A, const __bf16* __restrict__ B,
    const float* __restrict__ bias, float* Cf, __bf16* Cb,
    int M, int Nc, int K, long sB, long sC, int relu)
{
    __shared__ __bf16 As[GBM][LDSK];
    __shared__ __bf16 Bs[GBN][LDSK];   // transposed: Bs[n][k]

    const int tid  = threadIdx.x;
    const int m0   = blockIdx.x * GBM;
    const int n0   = blockIdx.y * GBN;
    const int z    = blockIdx.z;
    const __bf16* Bz = B + (long)z * sB;

    const int wave = tid >> 5, lane = tid & 31;
    const int wr   = wave & 3, wc = wave >> 2;     // row strip / col strip
    const int half = lane >> 4, ln = lane & 15;

    v8f acc[4];
    #pragma unroll
    for (int i = 0; i < 4; ++i)
        #pragma unroll
        for (int j = 0; j < 8; ++j) acc[i][j] = 0.0f;

    const int arow = tid >> 2;          // 0..63
    const int acol = (tid & 3) * 8;     // 0,8,16,24

#if USE_TDM
    const unsigned asLds = (unsigned)(size_t)(&As[0][0]);
#endif

    for (int k0 = 0; k0 < K; k0 += GBK) {
#if USE_TDM
        // ---- A tile (64x32) via Tensor Data Mover, wave 0 issues ----
        if (wave == 0) {
            tdm_load_2d_bf16(A + (long)m0 * K + k0, asLds,
                             (unsigned)(K - k0), (unsigned)(M - m0),
                             GBK, GBM, (unsigned)K,
                             /*pad_interval 16 DW=*/3u, /*pad 4 DW=*/3u);
        }
#else
        // ---- cooperative A tile load (64x32), one b128 per thread ----
        uint4 av = make_uint4(0u, 0u, 0u, 0u);
        if (m0 + arow < M)
            av = *(const uint4*)(A + (long)(m0 + arow) * K + k0 + acol);
        *(uint4*)(&As[arow][acol]) = av;
#endif

        // ---- cooperative B tile load (32x128) with transpose into LDS ----
        #pragma unroll
        for (int i = 0; i < 2; ++i) {
            int idx = tid + i * 256;
            int kk  = idx >> 4;            // 0..31
            int nn  = (idx & 15) * 8;      // 0..120
            uint4 bv = *(const uint4*)(Bz + (long)(k0 + kk) * Nc + n0 + nn);
            const __bf16* bpv = (const __bf16*)&bv;
            #pragma unroll
            for (int j = 0; j < 8; ++j) Bs[nn + j][kk] = bpv[j];
        }

#if USE_TDM
        if (wave == 0) __builtin_amdgcn_s_wait_tensorcnt(0);
#endif
        __syncthreads();

        // prefetch next B K-slab while we compute (global_prefetch_b8)
        if (k0 + GBK < K) {
            __builtin_prefetch(Bz + (long)(k0 + GBK + (tid >> 4)) * Nc + n0 + (tid & 15) * 8, 0, 0);
        }

        // ---- A fragment: 16x32 bf16, documented wave32 layout ----
        union { uint4 q[2]; v16bf v; } af;
        const __bf16* ap = &As[wr * 16 + ln][half * 8];
        af.q[0] = *(const uint4*)(ap);        // K = half*8 .. +7
        af.q[1] = *(const uint4*)(ap + 16);   // K = 16+half*8 .. +7

        // ---- 4 B fragments + WMMA ----
        #pragma unroll
        for (int tc = 0; tc < 4; ++tc) {
            union { uint4 q[2]; v16bf v; } bf_;
            const __bf16* bp = &Bs[wc * 64 + tc * 16 + ln][half * 8];
            bf_.q[0] = *(const uint4*)(bp);       // K = half*8 .. +7
            bf_.q[1] = *(const uint4*)(bp + 16);  // K = 16+half*8 .. +7
            acc[tc] = __builtin_amdgcn_wmma_f32_16x16x32_bf16(
                false, af.v, false, bf_.v, (short)0, acc[tc], false, false);
        }
        __syncthreads();
    }

    // ---- epilogue: D element r -> row half*8+r, col = lane&15 ----
    const int row_base = m0 + wr * 16 + half * 8;
    #pragma unroll
    for (int tc = 0; tc < 4; ++tc) {
        int col = n0 + wc * 64 + tc * 16 + ln;
        float bv = bias ? bias[col] : 0.0f;
        #pragma unroll
        for (int r = 0; r < 8; ++r) {
            int row = row_base + r;
            if (row < M) {
                float v = acc[tc][r] + bv;
                if (relu) v = fmaxf(v, 0.0f);
                long off = (long)z * sC + (long)row * Nc + col;
                if (Cb) Cb[off] = (__bf16)v;
                else    Cf[off] = v;
            }
        }
    }
}

// ---------------------------------------------------------------------------
// elementwise / helper kernels
// ---------------------------------------------------------------------------
__global__ __launch_bounds__(256) void k_cvt(const float* __restrict__ in,
                                             __bf16* __restrict__ out, long n) {
    for (long i = (long)blockIdx.x * blockDim.x + threadIdx.x; i < n;
         i += (long)gridDim.x * blockDim.x)
        out[i] = (__bf16)in[i];
}

// out[C,R] = in[R,C]^T  (bf16)
__global__ __launch_bounds__(256) void k_cvt_t(const float* __restrict__ in,
                                               __bf16* __restrict__ out, int R, int C) {
    long n = (long)R * C;
    for (long i = (long)blockIdx.x * blockDim.x + threadIdx.x; i < n;
         i += (long)gridDim.x * blockDim.x) {
        int r = (int)(i / C), c = (int)(i % C);
        out[(long)c * R + r] = (__bf16)in[i];
    }
}

__global__ __launch_bounds__(256) void k_zero(float* p, long n) {
    for (long i = (long)blockIdx.x * blockDim.x + threadIdx.x; i < n;
         i += (long)gridDim.x * blockDim.x)
        p[i] = 0.0f;
}

// s[h,n] = h(h,n,:) . a[h,0:D] ; t[h,n] = h(h,n,:) . a[h,D:2D]   (wave per row)
__global__ __launch_bounds__(256) void k_st(const __bf16* __restrict__ hb,
                                            const float* __restrict__ a,
                                            float* __restrict__ s,
                                            float* __restrict__ t) {
    const int lane = threadIdx.x & 31;
    long w      = (long)blockIdx.x * 8 + (threadIdx.x >> 5);
    long stride = (long)gridDim.x * 8;
    const long total = (long)HH * NN;
    for (; w < total; w += stride) {
        int h = (int)(w / NN);
        const __bf16* hp = hb + w * DD;
        const float*  al = a + (long)h * 2 * DD;
        int d = lane * 4;
        ushort4 u = *(const ushort4*)(hp + d);
        float f0 = b2f(u.x), f1 = b2f(u.y), f2 = b2f(u.z), f3 = b2f(u.w);
        float ss = f0 * al[d] + f1 * al[d + 1] + f2 * al[d + 2] + f3 * al[d + 3];
        float tt = f0 * al[DD + d] + f1 * al[DD + d + 1] +
                   f2 * al[DD + d + 2] + f3 * al[DD + d + 3];
        #pragma unroll
        for (int off = 16; off > 0; off >>= 1) {
            ss += __shfl_xor(ss, off, 32);
            tt += __shfl_xor(tt, off, 32);
        }
        if (lane == 0) { s[w] = ss; t[w] = tt; }
    }
}

// wave per (edge, head): att = exp(-leakyrelu(s[row]+t[col])); scatter-add
__global__ __launch_bounds__(256) void k_edge(const int* __restrict__ ei,
                                              const float* __restrict__ s,
                                              const float* __restrict__ t,
                                              const __bf16* __restrict__ hb,
                                              float* __restrict__ hp,
                                              float* __restrict__ rs) {
    const int lane = threadIdx.x & 31;
    long w      = (long)blockIdx.x * 8 + (threadIdx.x >> 5);
    long stride = (long)gridDim.x * 8;
    const long total = (long)EE * HH;
    for (; w < total; w += stride) {
        int e = (int)(w % EE);
        int h = (int)(w / EE);
        int row = ei[e], col = ei[EE + e];
        float ev = s[(long)h * NN + row] + t[(long)h * NN + col];
        float lr = ev > 0.0f ? ev : 0.2f * ev;
        float att = expf(-lr);
        if (lane == 0) atomicAdd(&rs[(long)h * NN + row], att);
        const __bf16* src = hb + ((long)h * NN + col) * DD + lane * 4;
        float*        dst = hp + ((long)h * NN + row) * DD + lane * 4;
        ushort4 u = *(const ushort4*)src;
        atomicAdd(dst + 0, att * b2f(u.x));
        atomicAdd(dst + 1, att * b2f(u.y));
        atomicAdd(dst + 2, att * b2f(u.z));
        atomicAdd(dst + 3, att * b2f(u.w));
    }
}

// out[n,d] = mean_h relu(hprime[h,n,d] / (rowsum[h,n] + 1e-5))  -> bf16
__global__ __launch_bounds__(256) void k_gatfin(const float* __restrict__ hp,
                                                const float* __restrict__ rs,
                                                __bf16* __restrict__ out) {
    const long total = (long)NN * DD;
    for (long i = (long)blockIdx.x * blockDim.x + threadIdx.x; i < total;
         i += (long)gridDim.x * blockDim.x) {
        int n = (int)(i / DD), d = (int)(i % DD);
        float acc = 0.0f;
        #pragma unroll
        for (int h = 0; h < HH; ++h) {
            float v = hp[((long)h * NN + n) * DD + d] /
                      (rs[(long)h * NN + n] + 1e-5f);
            acc += fmaxf(v, 0.0f);
        }
        out[i] = (__bf16)(acc * (1.0f / HH));
    }
}

// GRU pointwise gates; gi/gh include biases already (GEMM epilogue)
__global__ __launch_bounds__(256) void k_gate(const float* __restrict__ gi,
                                              const float* __restrict__ gh,
                                              float* __restrict__ h,
                                              __bf16* __restrict__ hb2) {
    const long total = (long)NN * DD;
    for (long i = (long)blockIdx.x * blockDim.x + threadIdx.x; i < total;
         i += (long)gridDim.x * blockDim.x) {
        int n = (int)(i / DD), d = (int)(i % DD);
        const float* gip = gi + (long)n * 3 * DD;
        const float* ghp = gh + (long)n * 3 * DD;
        float r  = sigmoidf_(gip[d] + ghp[d]);
        float z  = sigmoidf_(gip[DD + d] + ghp[DD + d]);
        float nn = tanhf(gip[2 * DD + d] + r * ghp[2 * DD + d]);
        float hv = h[i];
        float hn = (1.0f - z) * nn + z * hv;
        h[i] = hn;
        hb2[i] = (__bf16)hn;
    }
}

__global__ __launch_bounds__(256) void k_concat(const __bf16* __restrict__ a,
                                                const __bf16* __restrict__ b,
                                                __bf16* __restrict__ out) {
    const long total = (long)NN * DD;
    for (long i = (long)blockIdx.x * blockDim.x + threadIdx.x; i < total;
         i += (long)gridDim.x * blockDim.x) {
        int n = (int)(i / DD), d = (int)(i % DD);
        out[(long)n * 2 * DD + d]      = a[i];
        out[(long)n * 2 * DD + DD + d] = b[i];
    }
}

// logits[n,c] = x[n,:] . W[c,:] + b[c], c=0..2
__global__ __launch_bounds__(256) void k_head3(const __bf16* __restrict__ x,
                                               const float* __restrict__ W,
                                               const float* __restrict__ b,
                                               float* __restrict__ out) {
    for (long n = (long)blockIdx.x * blockDim.x + threadIdx.x; n < NN;
         n += (long)gridDim.x * blockDim.x) {
        float a0 = b[0], a1 = b[1], a2 = b[2];
        const __bf16* xp = x + n * DD;
        for (int d = 0; d < DD; ++d) {
            float xv = (float)xp[d];
            a0 += xv * W[d];
            a1 += xv * W[DD + d];
            a2 += xv * W[2 * DD + d];
        }
        out[n * 3 + 0] = a0;
        out[n * 3 + 1] = a1;
        out[n * 3 + 2] = a2;
    }
}

// ---------------------------------------------------------------------------
// host orchestration
// ---------------------------------------------------------------------------
static inline void gemm(hipStream_t st, const __bf16* A, const __bf16* B,
                        const float* bias, float* Cf, __bf16* Cb,
                        int M, int Nc, int K, long sB, long sC, int batch, int relu) {
    dim3 g((M + GBM - 1) / GBM, Nc / GBN, batch), b(256);
    k_gemm<<<g, b, 0, st>>>(A, B, bias, Cf, Cb, M, Nc, K, sB, sC, relu);
}

extern "C" void kernel_launch(void* const* d_in, const int* in_sizes, int n_in,
                              void* d_out, int out_size, void* d_ws, size_t ws_size,
                              hipStream_t stream) {
    const float* all_features = (const float*)d_in[0];
    const float* all_deg      = (const float*)d_in[1];
    const int*   edge_index   = (const int*)d_in[2];
    const float* gat1_W = (const float*)d_in[3];
    const float* gat1_a = (const float*)d_in[4];
    const float* gat2_W = (const float*)d_in[5];
    const float* gat2_a = (const float*)d_in[6];
    const float* gru_Wih = (const float*)d_in[7];
    const float* gru_Whh = (const float*)d_in[8];
    const float* gru_bih = (const float*)d_in[9];
    const float* gru_bhh = (const float*)d_in[10];
    const float* dgru_Wih = (const float*)d_in[11];
    const float* dgru_Whh = (const float*)d_in[12];
    const float* dgru_bih = (const float*)d_in[13];
    const float* dgru_bhh = (const float*)d_in[14];
    const float* fus_W = (const float*)d_in[15];
    const float* fus_b = (const float*)d_in[16];
    const float* pc1_W = (const float*)d_in[17];
    const float* pc1_b = (const float*)d_in[18];
    const float* pc2_W = (const float*)d_in[19];
    const float* pc2_b = (const float*)d_in[20];
    float* out = (float*)d_out;

    const size_t ND = (size_t)NN * DD;

    char* cur = (char*)d_ws;
    auto alloc = [&](size_t bytes) -> char* {
        char* r = cur;
        cur += (bytes + 255) & ~(size_t)255;
        return r;
    };

    __bf16* gat1b   = (__bf16*)alloc((size_t)HH * DD * DD * 2);
    __bf16* gat2b   = (__bf16*)alloc((size_t)HH * DD * DD * 2);
    __bf16* WihT    = (__bf16*)alloc((size_t)DD * 3 * DD * 2);
    __bf16* WhhT    = (__bf16*)alloc((size_t)DD * 3 * DD * 2);
    __bf16* dWihT   = (__bf16*)alloc((size_t)DD * 3 * DD * 2);
    __bf16* dWhhT   = (__bf16*)alloc((size_t)DD * 3 * DD * 2);
    __bf16* fusWT   = (__bf16*)alloc((size_t)2 * DD * DD * 2);
    __bf16* pc1WT   = (__bf16*)alloc((size_t)DD * DD * 2);
    __bf16* x0b     = (__bf16*)alloc(ND * 2);
    __bf16* x1b     = (__bf16*)alloc(ND * 2);
    __bf16* hb      = (__bf16*)alloc((size_t)HH * ND * 2);
    float*  sbuf    = (float*)alloc((size_t)HH * NN * 4);
    float*  tbuf    = (float*)alloc((size_t)HH * NN * 4);
    float*  hprime  = (float*)alloc(((size_t)HH * ND + (size_t)HH * NN) * 4);
    float*  rowsum  = hprime + (size_t)HH * ND;
    __bf16* featsb  = (__bf16*)alloc((size_t)TT * ND * 2);
    __bf16* degb    = (__bf16*)alloc((size_t)TT * ND * 2);
    float*  gi      = (float*)alloc((size_t)NN * 3 * DD * 4);
    float*  gh      = (float*)alloc((size_t)NN * 3 * DD * 4);
    float*  hseq    = (float*)alloc(ND * 4);
    float*  hdeg    = (float*)alloc(ND * 4);
    __bf16* hseqb   = (__bf16*)alloc(ND * 2);
    __bf16* hdegb   = (__bf16*)alloc(ND * 2);
    __bf16* concatb = (__bf16*)alloc(2 * ND * 2);
    __bf16* fusedb  = (__bf16*)alloc(ND * 2);
    __bf16* act1b   = (__bf16*)alloc(ND * 2);

    const dim3 B256(256);
    const dim3 Gbig(2048), Gmid(512), Gsml(64);

    // ---- weight conversions (bf16, GRU/fusion weights transposed) ----
    k_cvt  <<<Gsml, B256, 0, stream>>>(gat1_W, gat1b, (long)HH * DD * DD);
    k_cvt  <<<Gsml, B256, 0, stream>>>(gat2_W, gat2b, (long)HH * DD * DD);
    k_cvt_t<<<Gsml, B256, 0, stream>>>(gru_Wih,  WihT,  3 * DD, DD);
    k_cvt_t<<<Gsml, B256, 0, stream>>>(gru_Whh,  WhhT,  3 * DD, DD);
    k_cvt_t<<<Gsml, B256, 0, stream>>>(dgru_Wih, dWihT, 3 * DD, DD);
    k_cvt_t<<<Gsml, B256, 0, stream>>>(dgru_Whh, dWhhT, 3 * DD, DD);
    k_cvt_t<<<Gsml, B256, 0, stream>>>(fus_W, fusWT, DD, 2 * DD);
    k_cvt_t<<<Gsml, B256, 0, stream>>>(pc1_W, pc1WT, DD, DD);
    k_cvt  <<<Gbig, B256, 0, stream>>>(all_deg, degb, (long)TT * ND);

    // ---- per-timestep 2-layer sparse GAT ----
    for (int t = 0; t < TT; ++t) {
        const int* et = edge_index + (long)t * 2 * EE;
        k_cvt<<<Gbig, B256, 0, stream>>>(all_features + (long)t * ND, x0b, (long)ND);

        // layer 1: h[h] = x0 @ W1[h]  (batched over heads), bf16 out
        gemm(stream, x0b, gat1b, nullptr, nullptr, hb,
             NN, DD, DD, (long)DD * DD, (long)ND, HH, 0);
        k_st  <<<Gmid, B256, 0, stream>>>(hb, gat1_a, sbuf, tbuf);
        k_zero<<<Gbig, B256, 0, stream>>>(hprime, (long)HH * ND + (long)HH * NN);
        k_edge<<<Gbig, B256, 0, stream>>>(et, sbuf, tbuf, hb, hprime, rowsum);
        k_gatfin<<<Gbig, B256, 0, stream>>>(hprime, rowsum, x1b);

        // layer 2
        gemm(stream, x1b, gat2b, nullptr, nullptr, hb,
             NN, DD, DD, (long)DD * DD, (long)ND, HH, 0);
        k_st  <<<Gmid, B256, 0, stream>>>(hb, gat2_a, sbuf, tbuf);
        k_zero<<<Gbig, B256, 0, stream>>>(hprime, (long)HH * ND + (long)HH * NN);
        k_edge<<<Gbig, B256, 0, stream>>>(et, sbuf, tbuf, hb, hprime, rowsum);
        k_gatfin<<<Gbig, B256, 0, stream>>>(hprime, rowsum, featsb + (size_t)t * ND);
    }

    // ---- sequence GRU over feats ----
    k_zero<<<Gmid, B256, 0, stream>>>(hseq, (long)ND);
    k_zero<<<Gmid, B256, 0, stream>>>((float*)hseqb, (long)ND / 2);
    for (int t = 0; t < TT; ++t) {
        gemm(stream, featsb + (size_t)t * ND, WihT, gru_bih, gi, nullptr,
             NN, 3 * DD, DD, 0, 0, 1, 0);
        gemm(stream, hseqb, WhhT, gru_bhh, gh, nullptr,
             NN, 3 * DD, DD, 0, 0, 1, 0);
        k_gate<<<Gbig, B256, 0, stream>>>(gi, gh, hseq, hseqb);
    }

    // ---- degree GRU over all_deg ----
    k_zero<<<Gmid, B256, 0, stream>>>(hdeg, (long)ND);
    k_zero<<<Gmid, B256, 0, stream>>>((float*)hdegb, (long)ND / 2);
    for (int t = 0; t < TT; ++t) {
        gemm(stream, degb + (size_t)t * ND, dWihT, dgru_bih, gi, nullptr,
             NN, 3 * DD, DD, 0, 0, 1, 0);
        gemm(stream, hdegb, dWhhT, dgru_bhh, gh, nullptr,
             NN, 3 * DD, DD, 0, 0, 1, 0);
        k_gate<<<Gbig, B256, 0, stream>>>(gi, gh, hdeg, hdegb);
    }

    // ---- fusion + classifier ----
    k_concat<<<Gbig, B256, 0, stream>>>(hseqb, hdegb, concatb);
    gemm(stream, concatb, fusWT, fus_b, nullptr, fusedb,
         NN, DD, 2 * DD, 0, 0, 1, 0);
    gemm(stream, fusedb, pc1WT, pc1_b, nullptr, act1b,
         NN, DD, DD, 0, 0, 1, 1);
    k_head3<<<Gmid, B256, 0, stream>>>(act1b, pc2_W, pc2_b, out);
}
